// AttnDecoderRNN_8890582303037
// MI455X (gfx1250) — matrix operand order repaired and checked by reading, not compiled, not run on hardware
//
#include <hip/hip_runtime.h>
#include <math.h>

// ---------------------------------------------------------------------------
// AttnDecoderRNN for MI455X (gfx1250): bf16 WMMA GEMMs (64x128 tiles, TDM
// weight staging) + f32 VALU nonlinearities.
// B=64, S=128, T=64, H=512, V=32000.
// ---------------------------------------------------------------------------

typedef __bf16 bf16;
typedef __attribute__((ext_vector_type(16))) __bf16 v16bf;
typedef __attribute__((ext_vector_type(8)))  float  v8f;
typedef __attribute__((ext_vector_type(4)))  unsigned int u32x4;
typedef __attribute__((ext_vector_type(8)))  int          i32x8;
typedef __attribute__((ext_vector_type(4)))  int          i32x4;

constexpr int Bn = 64;
constexpr int Sn = 128;
constexpr int Tn = 64;
constexpr int Hn = 512;
constexpr int Vn = 32000;
constexpr int H2 = 2 * Hn;     // 1024
constexpr int H3 = 3 * Hn;     // 1536

// ---------------------------------------------------------------------------
// f32 -> bf16 conversion kernels
// ---------------------------------------------------------------------------
__global__ void cvt_f32_bf16(const float* __restrict__ in, bf16* __restrict__ out, int n) {
    int i = blockIdx.x * blockDim.x + threadIdx.x;
    if (i < n) out[i] = (bf16)in[i];
}

// out[c*rows + r] = in[r*cols + c]   (transpose + convert; weights become [N,K])
__global__ void cvtT_f32_bf16(const float* __restrict__ in, bf16* __restrict__ out,
                              int rows, int cols) {
    int i = blockIdx.x * blockDim.x + threadIdx.x;
    if (i < rows * cols) {
        int r = i / cols, c = i % cols;
        out[(size_t)c * rows + r] = (bf16)in[i];
    }
}

// ---------------------------------------------------------------------------
// bf16 NT GEMM: C[M,N] = A[M,K] * B[N,K]^T + bias[N], f32 accumulate.
// Block = 256 threads (8 waves), block tile 64(M) x 128(N), K step 32.
// Wave grid 2(M) x 4(N); each wave owns a 32x32 C sub-tile = 4 WMMA accs.
// B tile (128x32 bf16) is staged into LDS by the Tensor Data Mover
// (tensor_load_to_lds, TENSORcnt); A tile (64x32) by cooperative b128 loads.
// Requires: M%64==0, N%128==0, K%32==0 (true for all our shapes).
// ---------------------------------------------------------------------------
__global__ void gemm_bf16_nt(const bf16* __restrict__ A, int lda,
                             const bf16* __restrict__ Bw, int ldb,
                             float* __restrict__ C, int ldc,
                             const float* __restrict__ bias, int K, int N) {
    __shared__ __attribute__((aligned(16))) bf16 As[64 * 32];
    __shared__ __attribute__((aligned(16))) bf16 Bs[128 * 32];

    const int mBase = blockIdx.y * 64;
    const int nBase = blockIdx.x * 128;
    const int tid   = threadIdx.x;
    const int lane  = tid & 31;
    const int w     = tid >> 5;
    const int wm    = w >> 2;   // 0..1  (M sub-tile)
    const int wn    = w & 3;    // 0..3  (N sub-tile)

    v8f acc00 = {}, acc01 = {}, acc10 = {}, acc11 = {};

    // LDS byte offset of Bs: low 32 bits of the generic address.
    const unsigned ldsB = (unsigned)(unsigned long long)(const void*)&Bs[0];

    for (int kb = 0; kb < K; kb += 32) {
        __syncthreads();   // previous iteration's compute done before restage

        // ---- Stage A tile: 64x32 bf16, 8 elems (16B) per thread ----
        {
            int e = tid * 8;
            int r = e >> 5, c = e & 31;
            *reinterpret_cast<uint4*>(&As[r * 32 + c]) =
                *reinterpret_cast<const uint4*>(A + (size_t)(mBase + r) * lda + kb + c);
        }
        if (kb + 32 < K) {
            __builtin_prefetch(A + (size_t)(mBase + (tid & 63)) * lda + kb + 32, 0, 0);
        }

        // ---- Stage B tile via TDM: 2D tile 32(K) x 128(N rows) bf16 ----
        if (tid < 32) {
            unsigned long long gaddr =
                (unsigned long long)(const void*)(Bw + (size_t)nBase * ldb + kb);
            // Group 0: count=1 | lds_addr | global_addr | type=2
            u32x4 g0;
            g0[0] = 1u;                                    // count=1, user mode
            g0[1] = ldsB;                                  // lds_addr
            g0[2] = (unsigned)(gaddr & 0xffffffffu);       // global_addr[31:0]
            g0[3] = (unsigned)((gaddr >> 32) & 0x01ffffffu) | 0x80000000u; // [56:32] | type=2
            // Group 1: data_size=2B, tensor dims, tile 32 x 128, dim0 stride
            unsigned td0 = (unsigned)K;      // contiguous dim extent
            unsigned td1 = (unsigned)N;      // row count
            i32x8 g1;
            g1[0] = (int)0x00010000u;                      // data_size=1 (2 bytes)
            g1[1] = (int)((td0 & 0xffffu) << 16);          // tensor_dim0[15:0]
            g1[2] = (int)((td0 >> 16) | ((td1 & 0xffffu) << 16));
            g1[3] = (int)((td1 >> 16) | (32u << 16));      // tile_dim0 = 32
            g1[4] = (int)128;                              // tile_dim1 = 128, tile_dim2 = 0
            g1[5] = (int)(unsigned)ldb;                    // tensor_dim0_stride[31:0]
            g1[6] = 0;                                     // stride[47:32] | dim1_stride lo
            g1[7] = 0;
            i32x4 z4 = {0, 0, 0, 0};
            i32x8 z8 = {0, 0, 0, 0, 0, 0, 0, 0};
            __builtin_amdgcn_tensor_load_to_lds(g0, g1, z4, z4, z8, 0);
            __builtin_amdgcn_s_wait_tensorcnt(0);
        }
        __syncthreads();   // staged tiles visible to all waves

        // ---- Fragments (contiguous k = 16*(lane>>4) + j, same order for A and B) ----
        // A 16x32: lane<16 -> M=lane, K {0..15}; lane>=16 -> K {16..31}
        v16bf af0, af1, bf0, bf1;
        {
            int r0   = wm * 32 + (lane & 15);
            int koff = (lane >> 4) * 16;
#pragma unroll
            for (int j = 0; j < 8; ++j) af0[j]     = As[r0 * 32 + koff + j];
#pragma unroll
            for (int j = 0; j < 8; ++j) af0[8 + j] = As[r0 * 32 + 8 + koff + j];
#pragma unroll
            for (int j = 0; j < 8; ++j) af1[j]     = As[(r0 + 16) * 32 + koff + j];
#pragma unroll
            for (int j = 0; j < 8; ++j) af1[8 + j] = As[(r0 + 16) * 32 + 8 + koff + j];
        }
        // B 32x16 (K-major): lane&15 = N, (lane>>4)*16 + j = K
        {
            int n0   = wn * 32 + (lane & 15);
            int kofs = (lane >> 4) * 16;
#pragma unroll
            for (int j = 0; j < 16; ++j) bf0[j] = Bs[n0 * 32 + kofs + j];
#pragma unroll
            for (int j = 0; j < 16; ++j) bf1[j] = Bs[(n0 + 16) * 32 + kofs + j];
        }

        acc00 = __builtin_amdgcn_wmma_f32_16x16x32_bf16(false, af0, false, bf0, (short)0, acc00, false, false);
        acc01 = __builtin_amdgcn_wmma_f32_16x16x32_bf16(false, af0, false, bf1, (short)0, acc01, false, false);
        acc10 = __builtin_amdgcn_wmma_f32_16x16x32_bf16(false, af1, false, bf0, (short)0, acc10, false, false);
        acc11 = __builtin_amdgcn_wmma_f32_16x16x32_bf16(false, af1, false, bf1, (short)0, acc11, false, false);
    }

    // ---- Epilogue. C layout: N = lane&15, M = 8*(lane>>4) + vgpr_index ----
    const int nc = lane & 15;
    const int mc = (lane >> 4) * 8;
#pragma unroll
    for (int mi = 0; mi < 2; ++mi) {
#pragma unroll
        for (int ni = 0; ni < 2; ++ni) {
            const v8f* accp = (mi == 0) ? (ni == 0 ? &acc00 : &acc01)
                                        : (ni == 0 ? &acc10 : &acc11);
            int n  = nBase + wn * 32 + ni * 16 + nc;
            int m0 = mBase + wm * 32 + mi * 16 + mc;
            float bv = bias ? bias[n] : 0.0f;
#pragma unroll
            for (int j = 0; j < 8; ++j)
                C[(size_t)(m0 + j) * ldc + n] = (*accp)[j] + bv;
        }
    }
}

// ---------------------------------------------------------------------------
// Fused attention: scores -> softmax -> ctx; writes attention weights to d_out
// and builds bf16 x = [emb[inp], ctx] for the GRU input GEMM.
// One block per batch row b, 256 threads.
// ---------------------------------------------------------------------------
__global__ void attn_kernel(const float* __restrict__ hWa,    // [B,H] (h@Wa+ba)
                            const float* __restrict__ Uk,     // [B,S,H]
                            const float* __restrict__ Va,     // [H]
                            const float* __restrict__ bv,     // [1]
                            const float* __restrict__ enc,    // [B,S,H]
                            const float* __restrict__ emb,    // [V,H]
                            const int*   __restrict__ target, // [B,T]
                            int t,
                            float* __restrict__ attn_out,     // [B,T,S]
                            bf16*  __restrict__ x) {          // [B,2H]
    const int b   = blockIdx.x;
    const int tid = threadIdx.x;
    const int lane = tid & 31, w = tid >> 5;

    __shared__ float sH[Hn];
    __shared__ float sV[Hn];
    __shared__ float sS[Sn];
    __shared__ float sRed[8];

    for (int h = tid; h < Hn; h += 256) { sH[h] = hWa[b * Hn + h]; sV[h] = Va[h]; }
    __syncthreads();

    // scores[s] = Va . tanh(hWa + Uk[b,s,:]) + bv ; one wave per set of s.
    for (int s = w; s < Sn; s += 8) {
        const float* uk = Uk + ((size_t)b * Sn + s) * Hn;
        float acc = 0.0f;
        for (int h = lane; h < Hn; h += 32)
            acc += sV[h] * tanhf(sH[h] + uk[h]);
        for (int off = 16; off; off >>= 1) acc += __shfl_xor(acc, off, 32);
        if (lane == 0) sS[s] = acc + bv[0];
    }
    __syncthreads();

    // softmax over S=128
    float m = -INFINITY;
    for (int s = tid; s < Sn; s += 256) m = fmaxf(m, sS[s]);
    for (int off = 16; off; off >>= 1) m = fmaxf(m, __shfl_xor(m, off, 32));
    if (lane == 0) sRed[w] = m;
    __syncthreads();
    if (tid == 0) {
        float mm = sRed[0];
        for (int i = 1; i < 8; ++i) mm = fmaxf(mm, sRed[i]);
        sRed[0] = mm;
    }
    __syncthreads();
    m = sRed[0];
    __syncthreads();

    for (int s = tid; s < Sn; s += 256) sS[s] = __expf(sS[s] - m);
    __syncthreads();
    float sum = 0.0f;
    for (int s = tid; s < Sn; s += 256) sum += sS[s];
    for (int off = 16; off; off >>= 1) sum += __shfl_xor(sum, off, 32);
    if (lane == 0) sRed[w] = sum;
    __syncthreads();
    if (tid == 0) {
        float s2 = 0.0f;
        for (int i = 0; i < 8; ++i) s2 += sRed[i];
        sRed[0] = s2;
    }
    __syncthreads();
    const float inv = 1.0f / sRed[0];
    for (int s = tid; s < Sn; s += 256) {
        float ws = sS[s] * inv;
        sS[s] = ws;
        attn_out[((size_t)b * Tn + t) * Sn + s] = ws;
    }
    __syncthreads();

    // ctx = sum_s w[s] * enc[b,s,:] ; x = [emb[inp], ctx] in bf16.
    const int row = (t == 0) ? 0 : target[b * Tn + (t - 1)];
    for (int h = tid; h < Hn; h += 256) {
        float acc = 0.0f;
        for (int s = 0; s < Sn; ++s)
            acc += sS[s] * enc[((size_t)b * Sn + s) * Hn + h];
        x[(size_t)b * H2 + h]      = (bf16)emb[(size_t)row * Hn + h];
        x[(size_t)b * H2 + Hn + h] = (bf16)acc;
    }
}

// ---------------------------------------------------------------------------
// GRU cell elementwise (PyTorch gate order r,z,n). Writes f32 h and bf16 h.
// ---------------------------------------------------------------------------
__global__ void gru_kernel(const float* __restrict__ gi, const float* __restrict__ gh,
                           float* __restrict__ h, bf16* __restrict__ hB,
                           float* __restrict__ hF, int writeFinal) {
    int i = blockIdx.x * blockDim.x + threadIdx.x;
    if (i >= Bn * Hn) return;
    int b = i / Hn, hh = i % Hn;
    const float* gib = gi + (size_t)b * H3;
    const float* ghb = gh + (size_t)b * H3;
    float r = 1.0f / (1.0f + __expf(-(gib[hh] + ghb[hh])));
    float z = 1.0f / (1.0f + __expf(-(gib[Hn + hh] + ghb[Hn + hh])));
    float n = tanhf(gib[2 * Hn + hh] + r * ghb[2 * Hn + hh]);
    float hn = (1.0f - z) * n + z * h[i];
    h[i]  = hn;
    hB[i] = (bf16)hn;
    if (writeFinal) hF[i] = hn;
}

// ---------------------------------------------------------------------------
// log_softmax over V per batch row; writes into d_out log_probs[b,t,:].
// ---------------------------------------------------------------------------
__global__ void lsm_kernel(const float* __restrict__ logits, float* __restrict__ out, int t) {
    const int b = blockIdx.x;
    const int tid = threadIdx.x, lane = tid & 31, w = tid >> 5;
    __shared__ float sRed[8];
    const float* row = logits + (size_t)b * Vn;

    float m = -INFINITY;
    for (int v = tid; v < Vn; v += 256) m = fmaxf(m, row[v]);
    for (int off = 16; off; off >>= 1) m = fmaxf(m, __shfl_xor(m, off, 32));
    if (lane == 0) sRed[w] = m;
    __syncthreads();
    if (tid == 0) {
        float mm = sRed[0];
        for (int i = 1; i < 8; ++i) mm = fmaxf(mm, sRed[i]);
        sRed[0] = mm;
    }
    __syncthreads();
    m = sRed[0];
    __syncthreads();

    float sum = 0.0f;
    for (int v = tid; v < Vn; v += 256) sum += __expf(row[v] - m);
    for (int off = 16; off; off >>= 1) sum += __shfl_xor(sum, off, 32);
    if (lane == 0) sRed[w] = sum;
    __syncthreads();
    if (tid == 0) {
        float s2 = 0.0f;
        for (int i = 0; i < 8; ++i) s2 += sRed[i];
        sRed[0] = s2;
    }
    __syncthreads();
    const float lse = m + logf(sRed[0]);
    float* o = out + ((size_t)b * Tn + t) * Vn;
    for (int v = tid; v < Vn; v += 256) o[v] = row[v] - lse;
}

// ---------------------------------------------------------------------------
// Host orchestration (graph-capture safe: kernels + one async D2D copy).
// ---------------------------------------------------------------------------
extern "C" void kernel_launch(void* const* d_in, const int* in_sizes, int n_in,
                              void* d_out, int out_size, void* d_ws, size_t ws_size,
                              hipStream_t stream) {
    const float* enc  = (const float*)d_in[0];   // [B,S,H]
    const float* h0   = (const float*)d_in[1];   // [1,B,H]
    const int*   tgt  = (const int*)  d_in[2];   // [B,T]
    const float* emb  = (const float*)d_in[3];   // [V,H]
    const float* Wa   = (const float*)d_in[4];   // [H,H]
    const float* ba   = (const float*)d_in[5];
    const float* Ua   = (const float*)d_in[6];   // [H,H]
    const float* bu   = (const float*)d_in[7];
    const float* Va   = (const float*)d_in[8];   // [H,1]
    const float* bv   = (const float*)d_in[9];
    const float* W_ih = (const float*)d_in[10];  // [3H,2H]
    const float* W_hh = (const float*)d_in[11];  // [3H,H]
    const float* b_ih = (const float*)d_in[12];
    const float* b_hh = (const float*)d_in[13];
    const float* Wout = (const float*)d_in[14];  // [V,H]
    const float* bout = (const float*)d_in[15];

    float* out       = (float*)d_out;
    float* out_logp  = out;                                  // [B,T,V]
    float* out_hf    = out + (size_t)Bn * Tn * Vn;           // [1,B,H]
    float* out_attn  = out_hf + (size_t)Bn * Hn;             // [B,T,S]

    // Workspace carve-up (256B aligned slabs, ~73 MB total).
    char*  wsb = (char*)d_ws;
    size_t off = 0;
    auto carve = [&](size_t bytes) -> void* {
        void* p = wsb + off;
        off = (off + bytes + 255) & ~(size_t)255;
        return p;
    };
    bf16*  WoutB = (bf16*) carve((size_t)Vn * Hn * 2);
    bf16*  WihB  = (bf16*) carve((size_t)H3 * H2 * 2);
    bf16*  WhhB  = (bf16*) carve((size_t)H3 * Hn * 2);
    bf16*  WaT   = (bf16*) carve((size_t)Hn * Hn * 2);
    bf16*  UaT   = (bf16*) carve((size_t)Hn * Hn * 2);
    bf16*  encB  = (bf16*) carve((size_t)Bn * Sn * Hn * 2);
    float* Uk    = (float*)carve((size_t)Bn * Sn * Hn * 4);
    float* h     = (float*)carve((size_t)Bn * Hn * 4);
    bf16*  hB    = (bf16*) carve((size_t)Bn * Hn * 2);
    bf16*  x     = (bf16*) carve((size_t)Bn * H2 * 2);
    float* hWa   = (float*)carve((size_t)Bn * Hn * 4);
    float* gi    = (float*)carve((size_t)Bn * H3 * 4);
    float* gh    = (float*)carve((size_t)Bn * H3 * 4);
    float* lgts  = (float*)carve((size_t)Bn * Vn * 4);
    (void)ws_size; (void)in_sizes; (void)n_in; (void)out_size;

    const int THR = 256;
    auto blks = [](long n) { return (int)((n + 255) / 256); };

    // --- One-time weight conversion + Uk precompute ---
    cvt_f32_bf16<<<blks((long)Vn * Hn), THR, 0, stream>>>(Wout, WoutB, Vn * Hn);
    cvt_f32_bf16<<<blks((long)H3 * H2), THR, 0, stream>>>(W_ih, WihB, H3 * H2);
    cvt_f32_bf16<<<blks((long)H3 * Hn), THR, 0, stream>>>(W_hh, WhhB, H3 * Hn);
    cvtT_f32_bf16<<<blks((long)Hn * Hn), THR, 0, stream>>>(Wa, WaT, Hn, Hn);
    cvtT_f32_bf16<<<blks((long)Hn * Hn), THR, 0, stream>>>(Ua, UaT, Hn, Hn);
    cvt_f32_bf16<<<blks((long)Bn * Sn * Hn), THR, 0, stream>>>(enc, encB, Bn * Sn * Hn);

    (void)hipMemcpyAsync(h, h0, (size_t)Bn * Hn * 4, hipMemcpyDeviceToDevice, stream);
    cvt_f32_bf16<<<blks((long)Bn * Hn), THR, 0, stream>>>(h0, hB, Bn * Hn);

    // Uk = enc @ Ua + bu : M=B*S=8192, N=H, K=H
    gemm_bf16_nt<<<dim3(Hn / 128, (Bn * Sn) / 64), THR, 0, stream>>>(
        encB, Hn, UaT, Hn, Uk, Hn, bu, Hn, Hn);

    // --- Sequential decode ---
    for (int t = 0; t < Tn; ++t) {
        // hWa = h @ Wa + ba : M=B, N=H, K=H
        gemm_bf16_nt<<<dim3(Hn / 128, Bn / 64), THR, 0, stream>>>(
            hB, Hn, WaT, Hn, hWa, Hn, ba, Hn, Hn);

        attn_kernel<<<Bn, THR, 0, stream>>>(hWa, Uk, Va, bv, enc, emb, tgt, t,
                                            out_attn, x);

        // gi = x @ W_ih^T + b_ih : M=B, N=3H, K=2H
        gemm_bf16_nt<<<dim3(H3 / 128, Bn / 64), THR, 0, stream>>>(
            x, H2, WihB, H2, gi, H3, b_ih, H2, H3);
        // gh = h @ W_hh^T + b_hh : M=B, N=3H, K=H
        gemm_bf16_nt<<<dim3(H3 / 128, Bn / 64), THR, 0, stream>>>(
            hB, Hn, WhhB, Hn, gh, H3, b_hh, Hn, H3);

        gru_kernel<<<blks((long)Bn * Hn), THR, 0, stream>>>(
            gi, gh, h, hB, out_hf, (t == Tn - 1) ? 1 : 0);

        // logits = h_new @ Wout^T + bout : M=B, N=V, K=H
        gemm_bf16_nt<<<dim3(Vn / 128, Bn / 64), THR, 0, stream>>>(
            hB, Hn, WoutB, Hn, lgts, Vn, bout, Hn, Vn);

        lsm_kernel<<<Bn, THR, 0, stream>>>(lgts, out_logp, t);
    }
}
